// AttentionLayer_31138512896334
// MI455X (gfx1250) — matrix operand order repaired and checked by reading, not compiled, not run on hardware
//
#include <hip/hip_runtime.h>
#include <hip/hip_bf16.h>

typedef __attribute__((ext_vector_type(16))) _Float16 v16h;
typedef __attribute__((ext_vector_type(8)))  _Float16 v8h;
typedef __attribute__((ext_vector_type(4)))  _Float16 v4h;
typedef __attribute__((ext_vector_type(8)))  float    v8f;

#define C_   512
#define L_   16384
#define H_   8
#define D_   64
#define BL_  65536   // B*L = 4*16384
#define EPS_ 1e-6f

// gfx1250 async global->LDS path (guarded: falls back to plain copy if the
// toolchain doesn't expose the builtin, so compile never regresses).
#if defined(__gfx1250__) && __has_builtin(__builtin_amdgcn_global_load_async_to_lds_b128)
#define HAVE_ASYNC_LDS 1
#else
#define HAVE_ASYNC_LDS 0
#endif

#if HAVE_ASYNC_LDS
// Parameter type revealed by the round-2 diagnostic:
//   arg0: __attribute__((vector_size(16))) int  __device__(AS1) *
//   arg1: LDS-side pointer (AS3), same pointee
typedef int v4i_vs __attribute__((vector_size(16)));
typedef __attribute__((address_space(1))) v4i_vs* async_gptr_t;
typedef __attribute__((address_space(3))) v4i_vs* async_lptr_t;
#endif

__device__ __forceinline__ void wait_asynccnt0() {
#if defined(__gfx1250__)
#if __has_builtin(__builtin_amdgcn_s_wait_asynccnt)
  __builtin_amdgcn_s_wait_asynccnt(0);
#else
  asm volatile("s_wait_asynccnt 0x0" ::: "memory");
#endif
#endif
}

// ---------- WMMA helpers ----------
__device__ __forceinline__ v8f WMMA(v16h a, v16h b, v8f c) {
  // D = A(16x32 f16) * B(32x16 f16) + C(16x16 f32)
  return __builtin_amdgcn_wmma_f32_16x16x32_f16(false, a, false, b, (short)0, c, false, false);
}

// A fragment: per lane 16 contiguous halfs = row m, K 0..15 (lanes 0-15)
// or 16..31 (lanes 16-31). Caller pre-adds hi*16 to p.
__device__ __forceinline__ v16h ldA(const _Float16* p) {
  v8h a = *(const v8h*)(p);
  v8h b = *(const v8h*)(p + 8);
  v16h r;
#pragma unroll
  for (int i = 0; i < 8; ++i) { r[i] = a[i]; r[i + 8] = b[i]; }
  return r;
}

// B fragment: per lane 16 contiguous halfs (col n, K 0..15 or 16..31).
// Caller pre-adds hi*16 to p.
__device__ __forceinline__ v16h ldB16(const _Float16* p) {
  v8h a = *(const v8h*)(p);
  v8h b = *(const v8h*)(p + 8);
  v16h r;
#pragma unroll
  for (int i = 0; i < 8; ++i) { r[i] = a[i]; r[i + 8] = b[i]; }
  return r;
}

// Branchless elu(v)+1: exp of clamped value is always valid -> v_cndmask, no
// EXEC divergence around v_exp_f32.
__device__ __forceinline__ float fmap(float v) {
  float e = __expf(fminf(v, 0.0f));
  return (v > 0.0f) ? (v + 1.0f) : e;
}

// ---------- prep: Wt[m][n][k] = W_m[k][n] as f16 ----------
__global__ void prep_w(const float* __restrict__ Wq, const float* __restrict__ Wk,
                       const float* __restrict__ Wv, _Float16* __restrict__ Wt) {
  int idx = blockIdx.x * 256 + threadIdx.x;        // 0 .. 3*512*512-1
  int m   = idx >> 18;
  int rem = idx & ((1 << 18) - 1);
  int n   = rem >> 9;
  int k   = rem & 511;
  const float* W = (m == 0) ? Wq : (m == 1) ? Wk : Wv;
  Wt[(size_t)idx] = (_Float16)W[k * 512 + n];
}

// ---------- zero KV / Ksum accumulators (must run every launch) ----------
__global__ void zero_acc(float* __restrict__ KV, float* __restrict__ Ks) {
  int i = blockIdx.x * 256 + threadIdx.x;
  if (i < 32 * 64 * 64) KV[i] = 0.0f;
  if (i < 32 * 64)      Ks[i] = 0.0f;
}

// ---------- kernel 1: fused QKV projection + feature map ----------
// Grid: BL/128 blocks of 256 threads (8 waves). LDS: 128 x 520 halfs (130KB).
// x tile staged once (f32->f16) for the full K depth, then all 3*512 output
// columns are produced from it (x read exactly once from HBM).
__global__ void __launch_bounds__(256) qkv_kernel(
    const float* __restrict__ x, const _Float16* __restrict__ Wt,
    const float* __restrict__ bq, const float* __restrict__ bk, const float* __restrict__ bv,
    _Float16* __restrict__ Qh, _Float16* __restrict__ Kt, _Float16* __restrict__ Vt) {
  extern __shared__ _Float16 xs[];               // [128][520] (pad 8 -> conflict-free b128)
  const int m0  = blockIdx.x * 128;
  const int tid = threadIdx.x;

#pragma unroll 1
  for (int i = tid; i < 128 * 128; i += 256) {
    int row = i >> 7;
    int c4  = (i & 127) << 2;
    float4 g = *(const float4*)(x + (size_t)(m0 + row) * C_ + c4);
    v4h h4; h4[0] = (_Float16)g.x; h4[1] = (_Float16)g.y;
            h4[2] = (_Float16)g.z; h4[3] = (_Float16)g.w;
    *(v4h*)(xs + row * 520 + c4) = h4;
  }
  __syncthreads();

  const int wave = tid >> 5, lane = tid & 31, lm = lane & 15, hi = lane >> 4;
  const _Float16* aBase = xs + (wave * 16 + lm) * 520 + hi * 16;
  const int l0   = (m0 & (L_ - 1)) + wave * 16 + hi * 8;   // s position for transposed store
  const int bidx = m0 >> 14;                               // batch index

  for (int mat = 0; mat < 3; ++mat) {
    const _Float16* Wm  = Wt + ((size_t)mat << 18);
    const float*   bias = (mat == 0) ? bq : (mat == 1) ? bk : bv;
    for (int n0 = 0; n0 < C_; n0 += 64) {
      v8f acc[4] = {};
#pragma unroll 2
      for (int k0 = 0; k0 < C_; k0 += 32) {
        v16h a = ldA(aBase + k0);
#pragma unroll
        for (int t = 0; t < 4; ++t) {
          v16h b = ldB16(Wm + (size_t)(n0 + 16 * t + lm) * C_ + k0 + hi * 16);
          acc[t] = WMMA(a, b, acc[t]);
        }
      }
      if (mat == 0) {
        // Q: feature map, row-major store [BL][512]
        const int R = m0 + wave * 16 + hi * 8;
#pragma unroll
        for (int t = 0; t < 4; ++t) {
          int col = n0 + 16 * t + lm;
          float bb = bias[col];
#pragma unroll
          for (int r = 0; r < 8; ++r) {
            float v = fmap(acc[t][r] + bb);
            Qh[(size_t)(R + r) * C_ + col] = (_Float16)v;
          }
        }
      } else {
        // K (with feature map) / V: transposed store T[bh][d][s], one b128/lane/tile
        _Float16* T = (mat == 1) ? Kt : Vt;
        const int h = n0 >> 6;
#pragma unroll
        for (int t = 0; t < 4; ++t) {
          int d = 16 * t + lm;
          float bb = bias[n0 + d];
          v8h pk;
#pragma unroll
          for (int r = 0; r < 8; ++r) {
            float v = acc[t][r] + bb;
            if (mat == 1) v = fmap(v);
            pk[r] = (_Float16)v;
          }
          *(v8h*)(T + ((size_t)(bidx * H_ + h) * D_ + d) * L_ + l0) = pk;
        }
      }
    }
  }
}

// ---------- kernel 2: KV[d][v] = sum_s K[s,d]V[s,v], Ksum[d] = sum_s K[s,d] ----------
// Grid: 32 (b*h) * 8 (S-chunks of 2048), 128 threads (4 waves, wave w -> d rows 16w..).
// V chunk is staged once per WG into LDS with GLOBAL_LOAD_ASYNC_TO_LDS_B128
// (removes the 4x per-wave redundant B-fragment reads); K rows are wave-private
// and stream straight from global. Ksum comes free via an all-ones B fragment.
#define KV_CS 512   // staged s-chunk length (halfs)

__global__ void __launch_bounds__(128) kv_kernel(
    const _Float16* __restrict__ Kt, const _Float16* __restrict__ Vt,
    float* __restrict__ KV, float* __restrict__ Ks) {
  extern __shared__ _Float16 vs[];               // [64][520]
  const int bh = blockIdx.x >> 3;
  const int sc = blockIdx.x & 7;
  const int tid = threadIdx.x, wave = tid >> 5, lane = tid & 31, lm = lane & 15, hi = lane >> 4;
  const int d0 = wave * 16;
  const size_t base = (size_t)bh * D_ * L_;
  const _Float16* aP = Kt + base + (size_t)(d0 + lm) * L_ + sc * 2048 + hi * 16;

  v16h ones;
#pragma unroll
  for (int i = 0; i < 16; ++i) ones[i] = (_Float16)1.0f;

  v8f acc[4] = {};
  v8f ks = {};

  for (int cc = 0; cc < 2048; cc += KV_CS) {
    // stage Vt[64][cc..cc+KV_CS) -> LDS (64KB, 4096 16-byte transfers, 32/thread)
    const _Float16* gsrc = Vt + base + sc * 2048 + cc;
#pragma unroll 1
    for (int j = 0; j < 32; ++j) {
      int flat = tid + 128 * j;                  // 0..4095
      int r  = flat >> 6;
      int cb = (flat & 63) << 3;                 // half offset within row
      const _Float16* gp = gsrc + (size_t)r * L_ + cb;
      _Float16* lp = vs + r * 520 + cb;
#if HAVE_ASYNC_LDS
      __builtin_amdgcn_global_load_async_to_lds_b128(
          (async_gptr_t)(uintptr_t)gp,
          (async_lptr_t)(uint32_t)(uintptr_t)lp,
          0, 0);
#else
      *(v8h*)lp = *(const v8h*)gp;
#endif
    }
#if HAVE_ASYNC_LDS
    wait_asynccnt0();
#endif
    __syncthreads();

#pragma unroll 2
    for (int k0 = 0; k0 < KV_CS; k0 += 32) {
      v16h a = ldA(aP + cc + k0);
      ks = WMMA(a, ones, ks);
#pragma unroll
      for (int t = 0; t < 4; ++t) {
        v16h b = ldB16(vs + (16 * t + lm) * 520 + k0 + hi * 16);
        acc[t] = WMMA(a, b, acc[t]);
      }
    }
    __syncthreads();   // all waves done reading before next chunk overwrites
  }

  // store transposed: KV[bh][v][d] (so the final GEMM's B fragments are contiguous)
#pragma unroll
  for (int t = 0; t < 4; ++t) {
    int v = 16 * t + lm;
    float* dst = KV + (size_t)bh * 4096 + v * 64 + d0 + hi * 8;
#pragma unroll
    for (int r = 0; r < 8; ++r) atomicAdd(dst + r, acc[t][r]);
  }
  if (lm == 0) {  // all 16 columns of ks are identical; lanes 0/16 cover d0+{0..7}/{8..15}
    float* dst = Ks + bh * 64 + d0 + hi * 8;
#pragma unroll
    for (int r = 0; r < 8; ++r) atomicAdd(dst + r, ks[r]);
  }
}

// ---------- kernel 2.5: f32 accumulators -> f16 operands ----------
__global__ void cvt_kernel(const float* __restrict__ KV, const float* __restrict__ Ks,
                           _Float16* __restrict__ KVh, _Float16* __restrict__ Ksh) {
  int i = blockIdx.x * 256 + threadIdx.x;
  if (i < 32 * 64 * 64) KVh[i] = (_Float16)KV[i];
  if (i < 32 * 64)      Ksh[i] = (_Float16)Ks[i];
}

// ---------- kernel 3: out = (Q @ KV) / (Q . Ksum + eps) ----------
// Grid: (BL/128) * H blocks, 256 threads. Z denominator via Ksum-broadcast B fragment.
__global__ void __launch_bounds__(256) out_kernel(
    const _Float16* __restrict__ Qh, const _Float16* __restrict__ KVh,
    const _Float16* __restrict__ Ksh, float* __restrict__ out) {
  const int h  = blockIdx.x & 7;
  const int m0 = (blockIdx.x >> 3) * 128;
  const int tid = threadIdx.x, wave = tid >> 5, lane = tid & 31, lm = lane & 15, hi = lane >> 4;
  const int bh = (m0 >> 14) * H_ + h;
  const _Float16* aP = Qh + (size_t)(m0 + wave * 16 + lm) * C_ + h * D_ + hi * 16;
  const _Float16* bP = KVh + (size_t)bh * 4096 + hi * 16;
  const _Float16* zP = Ksh + bh * 64 + hi * 16;

  v8f acc[4] = {};
  v8f z = {};
#pragma unroll
  for (int k0 = 0; k0 < D_; k0 += 32) {
    v16h a  = ldA(aP + k0);
    v16h zb = ldB16(zP + k0);          // B[k][n] = Ksum[k] for all n
    z = WMMA(a, zb, z);
#pragma unroll
    for (int t = 0; t < 4; ++t) {
      v16h b = ldB16(bP + (16 * t + lm) * 64 + k0);
      acc[t] = WMMA(a, b, acc[t]);
    }
  }
  const int R = m0 + wave * 16 + hi * 8;
#pragma unroll
  for (int r = 0; r < 8; ++r) {
    float zi = 1.0f / (z[r] + EPS_);
#pragma unroll
    for (int t = 0; t < 4; ++t) {
      out[(size_t)(R + r) * C_ + h * D_ + 16 * t + lm] = acc[t][r] * zi;
    }
  }
}

extern "C" void kernel_launch(void* const* d_in, const int* in_sizes, int n_in,
                              void* d_out, int out_size, void* d_ws, size_t ws_size,
                              hipStream_t stream) {
  (void)in_sizes; (void)n_in; (void)out_size; (void)ws_size;
  const float* x  = (const float*)d_in[0];
  const float* Wq = (const float*)d_in[1];
  const float* bq = (const float*)d_in[2];
  const float* Wk = (const float*)d_in[3];
  const float* bk = (const float*)d_in[4];
  const float* Wv = (const float*)d_in[5];
  const float* bv = (const float*)d_in[6];
  float* out = (float*)d_out;

  char* ws = (char*)d_ws;
  _Float16* Qh  = (_Float16*)(ws);                                   // 64 MB
  _Float16* Kt  = (_Float16*)(ws + ((size_t)64 << 20));              // 64 MB
  _Float16* Vt  = (_Float16*)(ws + ((size_t)128 << 20));             // 64 MB
  _Float16* Wt  = (_Float16*)(ws + ((size_t)192 << 20));             // 1.5 MB
  float*    KV  = (float*)(ws + ((size_t)192 << 20) + 3 * 512 * 512 * 2);  // 512 KB
  float*    Ks  = KV + 32 * 64 * 64;                                 // 8 KB
  _Float16* KVh = (_Float16*)(Ks + 32 * 64);                         // 256 KB
  _Float16* Ksh = KVh + 32 * 64 * 64;                                // 4 KB

  prep_w  <<<3 * 512 * 512 / 256, 256, 0, stream>>>(Wq, Wk, Wv, Wt);
  zero_acc<<<512, 256, 0, stream>>>(KV, Ks);
  qkv_kernel<<<BL_ / 128, 256, 128 * 520 * sizeof(_Float16), stream>>>(
      x, Wt, bq, bk, bv, Qh, Kt, Vt);
  kv_kernel<<<32 * 8, 128, 64 * 520 * sizeof(_Float16), stream>>>(Kt, Vt, KV, Ks);
  cvt_kernel<<<512, 256, 0, stream>>>(KV, Ks, KVh, Ksh);
  out_kernel<<<(BL_ / 128) * H_, 256, 0, stream>>>(Qh, KVh, Ksh, out);
}
